// AttentionSequencePoolingLayer_13237089206375
// MI455X (gfx1250) — matrix operand order repaired and checked by reading, not compiled, not run on hardware
//
#include <hip/hip_runtime.h>

typedef __attribute__((ext_vector_type(16))) _Float16 v16h;
typedef __attribute__((ext_vector_type(8)))  float    v8f;
typedef __attribute__((ext_vector_type(4)))  int      v4i;
typedef __attribute__((address_space(1)))    v4i*     gv4p;   // global int4*
typedef __attribute__((address_space(3)))    v4i*     lv4p;   // LDS int4*

#define WAVES   4
#define KSTRIDE 68     // keys tile row stride in floats (bank-conflict-free, 16B aligned)
#define HSTRIDE 100    // hidden tile row stride in halves (bank-conflict-free, b32-aligned)

#if defined(__has_builtin)
#if __has_builtin(__builtin_amdgcn_global_load_async_to_lds_b128)
#define HAVE_ASYNC_LDS 1
#endif
#endif
#ifndef HAVE_ASYNC_LDS
#define HAVE_ASYNC_LDS 0
#endif

__device__ __forceinline__ float sigmoidf(float x) { return 1.f / (1.f + __expf(-x)); }

// ---------------------------------------------------------------------------
// Pack W0 (256x80) and W1 (zero-padded 96x48) into the CDNA5 WMMA B-fragment
// layout for v_wmma_f32_16x16x32_f16:
//   lane<16 : column n = lane,    half i -> K = i (i<8), 8 + i (i>=8)
//   lane>=16: column n = lane-16, half i -> K = 8 + i (i<8), 16 + i (i>=8)
// ---------------------------------------------------------------------------
__global__ void pack_weights_kernel(const float* __restrict__ W0,
                                    const float* __restrict__ W1,
                                    _Float16* __restrict__ wp)
{
    int idx = blockIdx.x * blockDim.x + threadIdx.x;
    if (idx < 40 * 512) {                       // W0: 8 K-chunks x 5 N-tiles
        int tile = idx >> 9, rem = idx & 511;
        int lane = rem >> 4, i = rem & 15;
        int kc = tile / 5, nt = tile % 5;
        int n = nt * 16 + (lane & 15);
        int k = kc * 32 + i + ((i >= 8) ? 8 : 0) + ((lane < 16) ? 0 : 8);
        wp[idx] = (_Float16)W0[k * 80 + n];
    } else if (idx < 49 * 512) {                // W1: 3 K-chunks x 3 N-tiles (padded)
        int idx2 = idx - 40 * 512;
        int tile = idx2 >> 9, rem = idx2 & 511;
        int lane = rem >> 4, i = rem & 15;
        int kc = tile / 3, nt = tile % 3;
        int n = nt * 16 + (lane & 15);
        int k = kc * 32 + i + ((i >= 8) ? 8 : 0) + ((lane < 16) ? 0 : 8);
        float v = (k < 80 && n < 40) ? W1[k * 40 + n] : 0.f;
        wp[idx] = (_Float16)v;
    }
}

// ---------------------------------------------------------------------------
// Main kernel: one wave per batch element, double-buffered async keys staging.
// ---------------------------------------------------------------------------
__global__ __launch_bounds__(128) void din_attention_kernel(
    const float* __restrict__ Q,      // [B,1,64]
    const float* __restrict__ K,      // [B,T,64]
    const int*   __restrict__ KLEN,   // [B,1]
    const float* __restrict__ B0,     // [80]
    const float* __restrict__ B1,     // [40]
    const float* __restrict__ B2,     // [1]
    const float* __restrict__ W2,     // [40,1]
    const _Float16* __restrict__ WP,  // packed W0 | W1 fragments
    float* __restrict__ OUT,          // [B,1,64]
    int T)
{
    __shared__ __align__(16) float    keysLds [WAVES][2][16 * KSTRIDE];
    __shared__                float    qLds    [WAVES][64];
    __shared__                _Float16 hLds    [WAVES][16 * HSTRIDE];
    __shared__                float    scoreLds[WAVES][16];

    const int w    = threadIdx.x >> 5;
    const int lane = threadIdx.x & 31;
    const int b    = blockIdx.x * WAVES + w;
    const int m    = lane & 15;               // column / row-in-tile id
    const int sA   = (lane < 16) ? 0 : 8;     // A-fragment / C-row half-wave shift
    const int row0 = lane >> 4;               // 0 or 1: staging row within pair
    const int col4 = (lane & 15) << 2;        // staging float4 column

    // stage this wave's query into LDS (wave-private region; same-wave DS in-order)
    qLds[w][lane]      = Q[b * 64 + lane];
    qLds[w][lane + 32] = Q[b * 64 + lane + 32];

    // zero the K-padding columns (80..99) of the hidden buffer once, branch-free
    #pragma unroll
    for (int j = 0; j < 10; ++j) {
        int pos = lane * 10 + j;
        int row = pos / 20, col = 80 + pos % 20;
        hLds[w][row * HSTRIDE + col] = (_Float16)0.f;
    }

    const int len = KLEN[b];

    float b0v[5], b1v[3], w2v[3];
    #pragma unroll
    for (int nt = 0; nt < 5; ++nt) b0v[nt] = B0[nt * 16 + m];
    #pragma unroll
    for (int nt = 0; nt < 3; ++nt) {
        int n2 = nt * 16 + m;
        b1v[nt] = (n2 < 40) ? B1[n2] : 0.f;
        w2v[nt] = (n2 < 40) ? W2[n2] : 0.f;
    }
    const float b2s = B2[0];

    // query A-fragments, built once (all rows identical)
    v16h aq[2];
    #pragma unroll
    for (int h = 0; h < 2; ++h) {
        const float* qr = &qLds[w][h * 32 + sA];
        #pragma unroll
        for (int i = 0; i < 16; ++i)
            aq[h][i] = (_Float16)qr[i + ((i >= 8) ? 8 : 0)];
    }

    const int ntiles = (T + 15) >> 4;
    // per-lane staging bases: tile stride = 16*64 floats, per-copy stride = 2 rows
    const float* gbase = K + ((size_t)b * T + row0) * 64 + col4;

    // Stage one 16x64 keys tile into keysLds[w][tile&1] (async, no VGPR round-trip).
    auto stage = [&](int tile) {
        float* lb = &keysLds[w][tile & 1][row0 * KSTRIDE + col4];
#if HAVE_ASYNC_LDS
        // WAR guard: prior reads of this buffer must have drained
        asm volatile("s_wait_dscnt 0x0" ::: "memory");
        if ((tile * 16 + 16) <= T) {
            const float* gp = gbase + (size_t)tile * 1024;
            #pragma unroll
            for (int it = 0; it < 8; ++it)
                __builtin_amdgcn_global_load_async_to_lds_b128(
                    (gv4p)(gp + it * 128), (lv4p)(lb + it * 2 * KSTRIDE), 0, 0);
        } else {  // ragged last tile: clamp rows; garbage rows are score-masked
            #pragma unroll
            for (int it = 0; it < 8; ++it) {
                int t = tile * 16 + it * 2 + row0;
                t = (t < T) ? t : (T - 1);
                const float* gp = K + ((size_t)b * T + t) * 64 + col4;
                __builtin_amdgcn_global_load_async_to_lds_b128(
                    (gv4p)gp, (lv4p)(lb + it * 2 * KSTRIDE), 0, 0);
            }
        }
#else
        #pragma unroll
        for (int g = 0; g < 2; ++g) {
            float4 tmp[4];
            #pragma unroll
            for (int j = 0; j < 4; ++j) {
                int t = tile * 16 + (g * 4 + j) * 2 + row0;
                t = (t < T) ? t : (T - 1);
                tmp[j] = *(const float4*)(K + ((size_t)b * T + t) * 64 + col4);
            }
            #pragma unroll
            for (int j = 0; j < 4; ++j)
                *(float4*)(lb + (g * 4 + j) * 2 * KSTRIDE) = tmp[j];
        }
#endif
    };

    float outAcc0 = 0.f, outAcc1 = 0.f;

    stage(0);
    for (int tile = 0; tile < ntiles; ++tile) {
        const int buf = tile & 1;
        if (tile + 1 < ntiles) stage(tile + 1);   // prefetch next tile (other buffer)
#if HAVE_ASYNC_LDS
        // async loads complete in order: <=8 outstanding => current tile landed
        if (tile + 1 < ntiles) asm volatile("s_wait_asynccnt 0x8" ::: "memory");
        else                   asm volatile("s_wait_asynccnt 0x0" ::: "memory");
#endif
        const float* ktile = keysLds[w][buf];

        // ---- key A-fragments (conflict-free gather: bank = (4m+e) mod 64) ----
        v16h ak[2];
        #pragma unroll
        for (int h = 0; h < 2; ++h) {
            const float* kr = &ktile[m * KSTRIDE + h * 32 + sA];
            #pragma unroll
            for (int i = 0; i < 16; ++i)
                ak[h][i] = (_Float16)kr[i + ((i >= 8) ? 8 : 0)];
        }

        // ---- layer 1: att(16x256) @ W0(256x80), 8 K-chunks x 5 N-tiles ----
        v8f acc1[5] = {};
        #pragma unroll
        for (int c = 0; c < 8; ++c) {
            const int h = c & 1, ty = c >> 1;
            v16h a;
            if      (ty == 0) a = aq[h];            // q
            else if (ty == 1) a = ak[h];            // k
            else if (ty == 2) a = aq[h] - ak[h];    // q-k (packed f16 sub)
            else              a = aq[h] * ak[h];    // q*k (packed f16 mul)
            #pragma unroll
            for (int n = 0; n < 5; ++n) {
                v16h bf = *(const v16h*)(WP + (size_t)(c * 5 + n) * 512 + lane * 16);
                acc1[n] = __builtin_amdgcn_wmma_f32_16x16x32_f16(
                    false, a, false, bf, (short)0, acc1[n], false, false);
            }
        }

        // ---- bias + sigmoid, write hidden (16x80) as f16 A-source in LDS ----
        #pragma unroll
        for (int n = 0; n < 5; ++n) {
            #pragma unroll
            for (int r = 0; r < 8; ++r) {
                float v = sigmoidf(acc1[n][r] + b0v[n]);
                hLds[w][(r + sA) * HSTRIDE + n * 16 + m] = (_Float16)v;
            }
        }

        // ---- layer 2: h(16x96) @ W1(96x48), 3 K-chunks x 3 N-tiles ----
        v8f acc2[3] = {};
        #pragma unroll
        for (int c = 0; c < 3; ++c) {
            v16h a;
            const _Float16* hr = &hLds[w][m * HSTRIDE + c * 32 + sA];
            #pragma unroll
            for (int i = 0; i < 16; ++i)
                a[i] = hr[i + ((i >= 8) ? 8 : 0)];
            #pragma unroll
            for (int n = 0; n < 3; ++n) {
                v16h bf = *(const v16h*)(WP + (size_t)(40 * 512) +
                                         (size_t)(c * 3 + n) * 512 + lane * 16);
                acc2[n] = __builtin_amdgcn_wmma_f32_16x16x32_f16(
                    false, a, false, bf, (short)0, acc2[n], false, false);
            }
        }

        // ---- layer 3 (N=1): lane-shuffle dot over the 16-lane column group ----
        #pragma unroll
        for (int r = 0; r < 8; ++r) {
            float s = 0.f;
            #pragma unroll
            for (int nt = 0; nt < 3; ++nt)
                s += sigmoidf(acc2[nt][r] + b1v[nt]) * w2v[nt];
            #pragma unroll
            for (int off = 1; off < 16; off <<= 1)
                s += __shfl_xor(s, off, 32);
            if (m == 0) {                       // lanes 0 (row r) and 16 (row r+8)
                int t = tile * 16 + r + sA;
                scoreLds[w][r + sA] = (t < len) ? sigmoidf(s + b2s) : 0.f;
            }
        }

        // ---- masked weighted sum: out[e] += score[t] * keys[t][e] ----
        #pragma unroll
        for (int mm = 0; mm < 16; ++mm) {
            float sm = scoreLds[w][mm];
            outAcc0 = fmaf(sm, ktile[mm * KSTRIDE + lane],      outAcc0);
            outAcc1 = fmaf(sm, ktile[mm * KSTRIDE + 32 + lane], outAcc1);
        }
    }

    OUT[b * 64 + lane]      = outAcc0;
    OUT[b * 64 + lane + 32] = outAcc1;
}

// ---------------------------------------------------------------------------
extern "C" void kernel_launch(void* const* d_in, const int* in_sizes, int n_in,
                              void* d_out, int out_size, void* d_ws, size_t ws_size,
                              hipStream_t stream)
{
    const float* queries = (const float*)d_in[0];
    const float* keys    = (const float*)d_in[1];
    const int*   klen    = (const int*)  d_in[2];
    const float* W0      = (const float*)d_in[3];
    const float* b0      = (const float*)d_in[4];
    const float* W1      = (const float*)d_in[5];
    const float* b1      = (const float*)d_in[6];
    const float* W2      = (const float*)d_in[7];
    const float* b2      = (const float*)d_in[8];
    float*       out     = (float*)d_out;

    const int B = in_sizes[0] / 64;                 // queries: B*1*64
    const int T = in_sizes[1] / (B * 64);           // keys:    B*T*64

    _Float16* wp = (_Float16*)d_ws;                 // 49*512 halves = 50176 B

    pack_weights_kernel<<<(49 * 512 + 255) / 256, 256, 0, stream>>>(W0, W1, wp);
    din_attention_kernel<<<B / WAVES, WAVES * 32, 0, stream>>>(
        queries, keys, klen, b0, b1, b2, W2, wp, out, T);
}
